// TransformerLayerShard_21775484190933
// MI455X (gfx1250) — matrix operand order repaired and checked by reading, not compiled, not run on hardware
//
#include <hip/hip_runtime.h>
#include <hip/hip_bf16.h>
#include <math.h>

typedef _Float16 v16h __attribute__((ext_vector_type(16)));
typedef float    v8f  __attribute__((ext_vector_type(8)));

#define T_SEQ  2048
#define DMODEL 4096
#define NHEAD  16
#define DHEAD  256
#define DFF    16384
#define ROTD   64

// ---------- fragment helpers ----------
__device__ __forceinline__ v16h frag_2x8(const _Float16* p0, const _Float16* p1) {
  v16h r;
#pragma unroll
  for (int i = 0; i < 8; i++) { r[i] = p0[i]; r[i + 8] = p1[i]; }
  return r;
}
__device__ __forceinline__ v16h frag_16(const _Float16* p) {
  v16h r;
#pragma unroll
  for (int i = 0; i < 16; i++) r[i] = p[i];
  return r;
}
__device__ __forceinline__ v8f wmma_f16(v16h a, v16h b, v8f c) {
  return __builtin_amdgcn_wmma_f32_16x16x32_f16(false, a, false, b, (short)0, c, false, false);
}

// ---------- fp32 -> f16 conversion ----------
__global__ void cvt_f32_to_f16(const float* __restrict__ in, _Float16* __restrict__ out, size_t n) {
  size_t i = (size_t)blockIdx.x * blockDim.x + threadIdx.x;
  size_t stride = (size_t)gridDim.x * blockDim.x;
  for (; i < n; i += stride) out[i] = (_Float16)in[i];
}

// ---------- LayerNorm (one row per block) ----------
__global__ void __launch_bounds__(256)
layernorm_f16(const float* __restrict__ x, const float* __restrict__ scale,
              const float* __restrict__ offset, _Float16* __restrict__ h16) {
  __shared__ float red[256];
  int row = blockIdx.x;
  const float* xr = x + (size_t)row * DMODEL;
  float s = 0.f;
  for (int i = threadIdx.x; i < DMODEL; i += 256) s += xr[i];
  red[threadIdx.x] = s; __syncthreads();
  for (int o = 128; o > 0; o >>= 1) { if (threadIdx.x < o) red[threadIdx.x] += red[threadIdx.x + o]; __syncthreads(); }
  float mean = red[0] / (float)DMODEL; __syncthreads();
  float v = 0.f;
  for (int i = threadIdx.x; i < DMODEL; i += 256) { float d = xr[i] - mean; v += d * d; }
  red[threadIdx.x] = v; __syncthreads();
  for (int o = 128; o > 0; o >>= 1) { if (threadIdx.x < o) red[threadIdx.x] += red[threadIdx.x + o]; __syncthreads(); }
  float rstd = rsqrtf(red[0] / (float)DMODEL + 1e-6f);
  for (int i = threadIdx.x; i < DMODEL; i += 256)
    h16[(size_t)row * DMODEL + i] = (_Float16)((xr[i] - mean) * rstd * scale[i] + offset[i]);
}

// ---------- WMMA GEMM: C[M,N] = A[M,K] @ B[K,N], f16 in, f32 accum ----------
// 128x128 block tile, BK=64, double-buffered LDS, 8 waves x (16x128) each.
// B fragments preloaded into distinct registers so ds_loads pipeline past WMMAs.
// mode 0: store f32   mode 1: +bias, gelu, store f16   mode 2: +bias, store f32   mode 3: store f16
#define ASTR 72
#define BSTR 136
__global__ void __launch_bounds__(256)
gemm_f16_wmma(const _Float16* __restrict__ A, const _Float16* __restrict__ B,
              int N, int K, const float* __restrict__ bias,
              float* __restrict__ outF, _Float16* __restrict__ outH, int mode) {
  const int BM = 128, BN = 128, BK = 64;
  __shared__ _Float16 As[2][128 * ASTR];
  __shared__ _Float16 Bs[2][64 * BSTR];
  int tid = threadIdx.x;
  int lane = tid & 31;
  int wid = tid >> 5;               // 0..7 -> wave owns rows wid*16..wid*16+15
  int bm = blockIdx.y * BM;
  int bn = blockIdx.x * BN;

  v8f c[8];
#pragma unroll
  for (int t = 0; t < 8; t++)
#pragma unroll
    for (int r = 0; r < 8; r++) c[t][r] = 0.f;

  int m = lane & 15;
  int hk = (lane >> 4) * 8;         // A-frag K-half select

  // staging maps: A 128x64 (4 chunks of 8/thread), B 64x128 (2 chunks of 16/thread)
  auto stage = [&](int s, int k0) {
#pragma unroll
    for (int cch = 0; cch < 4; cch++) {
      int idx = tid + cch * 256;
      int row = idx >> 3;
      int col = (idx & 7) * 8;
#pragma unroll
      for (int i = 0; i < 8; i++)
        As[s][row * ASTR + col + i] = A[(size_t)(bm + row) * K + k0 + col + i];
    }
#pragma unroll
    for (int cch = 0; cch < 2; cch++) {
      int idx = tid + cch * 256;
      int row = idx >> 3;
      int col = (idx & 7) * 16;
#pragma unroll
      for (int i = 0; i < 16; i++)
        Bs[s][row * BSTR + col + i] = B[(size_t)(k0 + row) * N + bn + col + i];
    }
  };

  stage(0, 0);
  __syncthreads();

  int nK = K / BK;
  for (int kb = 0; kb < nK; kb++) {
    int s = kb & 1;
    if (kb + 1 < nK) {
      stage(s ^ 1, (kb + 1) * BK);
      __builtin_prefetch(&A[(size_t)(bm + (tid >> 3)) * K + (kb + 2) * BK], 0, 1);
      __builtin_prefetch(&B[(size_t)((kb + 2) * BK + (tid >> 3)) * N + bn], 0, 1);
    }
    // compute on buffer s: two k=32 steps; preload 8 B fragments, then 8 WMMAs
#pragma unroll
    for (int kk = 0; kk < 2; kk++) {
      v16h a = frag_2x8(&As[s][(wid * 16 + m) * ASTR + kk * 32 + hk],
                        &As[s][(wid * 16 + m) * ASTR + kk * 32 + hk + 16]);
      v16h bfr[8];
#pragma unroll
      for (int t = 0; t < 8; t++)
#pragma unroll
        for (int i = 0; i < 16; i++)
          bfr[t][i] = Bs[s][(kk * 32 + hk + (i & 7) + (i >> 3) * 16) * BSTR + t * 16 + m];
#pragma unroll
      for (int t = 0; t < 8; t++)
        c[t] = wmma_f16(a, bfr[t], c[t]);
    }
    __syncthreads();
  }

#pragma unroll
  for (int t = 0; t < 8; t++) {
#pragma unroll
    for (int r = 0; r < 8; r++) {
      int row = bm + wid * 16 + r + ((lane >> 4) << 3);
      int col = bn + t * 16 + (lane & 15);
      float v = c[t][r];
      if (mode == 1 || mode == 2) v += bias[col];
      if (mode == 1) {
        float u = 0.7978845608028654f * (v + 0.044715f * v * v * v);
        v = 0.5f * v * (1.0f + tanhf(u));
      }
      size_t idx = (size_t)row * N + col;
      if (mode == 0 || mode == 2) outF[idx] = v;
      else                        outH[idx] = (_Float16)v;
    }
  }
}

// ---------- RoPE: q -> q16 (row major), k -> kT16 (H, Dh, T) ----------
__global__ void __launch_bounds__(256)
rope_qk(const float* __restrict__ q, const float* __restrict__ k,
        _Float16* __restrict__ q16, _Float16* __restrict__ kT16) {
  int t = blockIdx.x;
  for (int idx = threadIdx.x; idx < DMODEL; idx += 256) {
    int head = idx >> 8;
    int d = idx & 255;
    const float* qh = q + (size_t)t * DMODEL + head * DHEAD;
    const float* kh = k + (size_t)t * DMODEL + head * DHEAD;
    float qv, kv;
    if (d < ROTD) {
      int f = (d < 32) ? d : d - 32;
      float inv = __powf(10000.0f, -(float)f / 32.0f);
      float ang = (float)t * inv;
      float cs = __cosf(ang), sn = __sinf(ang);
      if (d < 32) { qv = qh[d] * cs - qh[d + 32] * sn;  kv = kh[d] * cs - kh[d + 32] * sn; }
      else        { qv = qh[d - 32] * sn + qh[d] * cs;  kv = kh[d - 32] * sn + kh[d] * cs; }
    } else { qv = qh[d]; kv = kh[d]; }
    q16[(size_t)t * DMODEL + idx] = (_Float16)qv;
    kT16[((size_t)head * DHEAD + d) * T_SEQ + t] = (_Float16)kv;
  }
}

// ---------- Flash attention: 4 waves/block, each wave owns one 16-query tile ----------
// No cross-wave barriers: each wave uses a private LDS slice; LDS ops are in-order per wave.
__global__ void __launch_bounds__(128)
flash_attn(const _Float16* __restrict__ q16, const _Float16* __restrict__ kT16,
           const _Float16* __restrict__ vsrc, _Float16* __restrict__ y16) {
  const int PSTR = 40;
  __shared__ _Float16 Ps[4][16 * PSTR];
  int lane = threadIdx.x & 31;
  int wid  = threadIdx.x >> 5;      // 0..3
  int qi = blockIdx.x * 4 + wid;    // query tile (16 rows)
  int h  = blockIdx.y;
  _Float16* myPs = &Ps[wid][0];
  int m = lane & 15;
  int hk = (lane >> 4) * 8;
  int rowOff = (lane >> 4) * 8;     // C-layout: lanes 16-31 hold rows 8..15

  // Q fragments for full Dh=256 (8 x 16x32 A-fragments), resident in VGPRs
  v16h qf[8];
  {
    const _Float16* qbase = q16 + (size_t)(qi * 16 + m) * DMODEL + h * DHEAD;
#pragma unroll
    for (int kk = 0; kk < 8; kk++)
      qf[kk] = frag_2x8(qbase + kk * 32 + hk, qbase + kk * 32 + hk + 16);
  }

  v8f o[16];
#pragma unroll
  for (int dt = 0; dt < 16; dt++)
#pragma unroll
    for (int r = 0; r < 8; r++) o[dt][r] = 0.f;
  float mrow[8], lrow[8];
#pragma unroll
  for (int r = 0; r < 8; r++) { mrow[r] = -3.0e38f; lrow[r] = 0.f; }

  int nb = (qi * 16 + 16 + 31) / 32;   // causal: key blocks of 32
  const float scale = 0.0625f;          // 1/sqrt(256)
  const _Float16* kThb = kT16 + (size_t)h * DHEAD * T_SEQ;

  for (int j = 0; j < nb; j++) {
    int kb = j * 32;
    v8f s0, s1;
#pragma unroll
    for (int r = 0; r < 8; r++) { s0[r] = 0.f; s1[r] = 0.f; }
    // S = Q @ K^T : preload 4 K-fragments (2 k-steps), then 4 WMMAs
#pragma unroll
    for (int kk = 0; kk < 8; kk += 2) {
      v16h b00, b01, b10, b11;
#pragma unroll
      for (int i = 0; i < 16; i++) {
        int kr = hk + (i & 7) + (i >> 3) * 16;
        b00[i] = kThb[(size_t)(kk * 32 + kr) * T_SEQ + kb + m];
        b01[i] = kThb[(size_t)(kk * 32 + kr) * T_SEQ + kb + 16 + m];
        b10[i] = kThb[(size_t)((kk + 1) * 32 + kr) * T_SEQ + kb + m];
        b11[i] = kThb[(size_t)((kk + 1) * 32 + kr) * T_SEQ + kb + 16 + m];
      }
      s0 = wmma_f16(qf[kk], b00, s0);
      s1 = wmma_f16(qf[kk], b01, s1);
      s0 = wmma_f16(qf[kk + 1], b10, s0);
      s1 = wmma_f16(qf[kk + 1], b11, s1);
    }
    // scale + causal mask
#pragma unroll
    for (int r = 0; r < 8; r++) {
      int rg = qi * 16 + r + rowOff;
      int n0 = kb + (lane & 15);
      s0[r] = s0[r] * scale + ((n0 > rg) ? -1.0e10f : 0.f);
      s1[r] = s1[r] * scale + ((n0 + 16 > rg) ? -1.0e10f : 0.f);
    }
    // online softmax (half-wave reductions: rows live in lane halves)
    float alpha[8];
    v8f p0, p1;
#pragma unroll
    for (int r = 0; r < 8; r++) {
      float mx = fmaxf(s0[r], s1[r]);
#pragma unroll
      for (int off = 1; off < 16; off <<= 1)
        mx = fmaxf(mx, __shfl_xor(mx, off, 32));
      float mn = fmaxf(mrow[r], mx);
      alpha[r] = __expf(mrow[r] - mn);
      mrow[r] = mn;
      p0[r] = __expf(s0[r] - mn);
      p1[r] = __expf(s1[r] - mn);
      float sm = p0[r] + p1[r];
#pragma unroll
      for (int off = 1; off < 16; off <<= 1)
        sm += __shfl_xor(sm, off, 32);
      lrow[r] = lrow[r] * alpha[r] + sm;
    }
#pragma unroll
    for (int dt = 0; dt < 16; dt++)
#pragma unroll
      for (int r = 0; r < 8; r++) o[dt][r] *= alpha[r];

    // P: C-layout -> private LDS slice -> A-fragment layout (in-order per wave)
#pragma unroll
    for (int r = 0; r < 8; r++) {
      int rl = r + rowOff;
      myPs[rl * PSTR + (lane & 15)]      = (_Float16)p0[r];
      myPs[rl * PSTR + 16 + (lane & 15)] = (_Float16)p1[r];
    }
    asm volatile("" ::: "memory");
    v16h pf = frag_2x8(&myPs[m * PSTR + hk], &myPs[m * PSTR + hk + 16]);

    // O += P @ V : preload V fragments in chunks of 4, then 4 WMMAs
#pragma unroll
    for (int dt0 = 0; dt0 < 16; dt0 += 4) {
      v16h vf[4];
#pragma unroll
      for (int i = 0; i < 4; i++)
#pragma unroll
        for (int e = 0; e < 16; e++)
          vf[i][e] = vsrc[(size_t)(kb + hk + (e & 7) + (e >> 3) * 16) * DMODEL + h * DHEAD + (dt0 + i) * 16 + m];
#pragma unroll
      for (int i = 0; i < 4; i++)
        o[dt0 + i] = wmma_f16(pf, vf[i], o[dt0 + i]);
    }
  }
  // y = O / l
#pragma unroll
  for (int dt = 0; dt < 16; dt++) {
#pragma unroll
    for (int r = 0; r < 8; r++) {
      int rg = qi * 16 + r + rowOff;
      int col = h * DHEAD + dt * 16 + (lane & 15);
      y16[(size_t)rg * DMODEL + col] = (_Float16)(o[dt][r] / lrow[r]);
    }
  }
}

// ---------- residual add ----------
__global__ void final_add(const float* __restrict__ x, const float* __restrict__ attn,
                          const float* __restrict__ mlp, float* __restrict__ out, size_t n) {
  size_t i = (size_t)blockIdx.x * blockDim.x + threadIdx.x;
  size_t stride = (size_t)gridDim.x * blockDim.x;
  for (; i < n; i += stride) out[i] = x[i] + attn[i] + mlp[i];
}

extern "C" void kernel_launch(void* const* d_in, const int* in_sizes, int n_in,
                              void* d_out, int out_size, void* d_ws, size_t ws_size,
                              hipStream_t stream) {
  const float* x     = (const float*)d_in[0];
  const float* wq    = (const float*)d_in[1];
  const float* wk    = (const float*)d_in[2];
  const float* wv    = (const float*)d_in[3];
  const float* wo    = (const float*)d_in[4];
  const float* w_in  = (const float*)d_in[5];
  const float* b_in  = (const float*)d_in[6];
  const float* w_out = (const float*)d_in[7];
  const float* b_out = (const float*)d_in[8];
  const float* lns   = (const float*)d_in[9];
  const float* lno   = (const float*)d_in[10];
  float* out = (float*)d_out;

  char* ws = (char*)d_ws;
  size_t off = 0;
  auto alloc = [&](size_t bytes) -> char* {
    char* p = ws + off; off += (bytes + 255) & ~(size_t)255; return p;
  };
  const size_t TD = (size_t)T_SEQ * DMODEL;
  _Float16* h16    = (_Float16*)alloc(TD * 2);
  _Float16* wq16   = (_Float16*)alloc((size_t)DMODEL * DMODEL * 2);
  _Float16* wk16   = (_Float16*)alloc((size_t)DMODEL * DMODEL * 2);
  _Float16* wv16   = (_Float16*)alloc((size_t)DMODEL * DMODEL * 2);
  _Float16* wo16   = (_Float16*)alloc((size_t)DMODEL * DMODEL * 2);
  _Float16* win16  = (_Float16*)alloc((size_t)DMODEL * DFF * 2);
  _Float16* wout16 = (_Float16*)alloc((size_t)DFF * DMODEL * 2);
  float*    qf     = (float*)   alloc(TD * 4);
  float*    kf     = (float*)   alloc(TD * 4);
  _Float16* v16b   = (_Float16*)alloc(TD * 2);
  _Float16* q16    = (_Float16*)alloc(TD * 2);
  _Float16* kT16   = (_Float16*)alloc(TD * 2);
  _Float16* y16    = (_Float16*)alloc(TD * 2);
  float*    attnF  = (float*)   alloc(TD * 4);
  _Float16* mh16   = (_Float16*)alloc((size_t)T_SEQ * DFF * 2);
  float*    mlpF   = (float*)   alloc(TD * 4);

  // weight conversion fp32 -> f16
  cvt_f32_to_f16<<<4096, 256, 0, stream>>>(wq,    wq16,   (size_t)DMODEL * DMODEL);
  cvt_f32_to_f16<<<4096, 256, 0, stream>>>(wk,    wk16,   (size_t)DMODEL * DMODEL);
  cvt_f32_to_f16<<<4096, 256, 0, stream>>>(wv,    wv16,   (size_t)DMODEL * DMODEL);
  cvt_f32_to_f16<<<4096, 256, 0, stream>>>(wo,    wo16,   (size_t)DMODEL * DMODEL);
  cvt_f32_to_f16<<<8192, 256, 0, stream>>>(w_in,  win16,  (size_t)DMODEL * DFF);
  cvt_f32_to_f16<<<8192, 256, 0, stream>>>(w_out, wout16, (size_t)DFF * DMODEL);

  // layernorm
  layernorm_f16<<<T_SEQ, 256, 0, stream>>>(x, lns, lno, h16);

  // q, k (fp32 for rope), v (f16 direct)
  dim3 gD(DMODEL / 128, T_SEQ / 128);   // (32, 16)
  gemm_f16_wmma<<<gD, 256, 0, stream>>>(h16, wq16, DMODEL, DMODEL, nullptr, qf, nullptr, 0);
  gemm_f16_wmma<<<gD, 256, 0, stream>>>(h16, wk16, DMODEL, DMODEL, nullptr, kf, nullptr, 0);
  gemm_f16_wmma<<<gD, 256, 0, stream>>>(h16, wv16, DMODEL, DMODEL, nullptr, nullptr, v16b, 3);

  // rope -> q16 row-major, kT16 transposed per head
  rope_qk<<<T_SEQ, 256, 0, stream>>>(qf, kf, q16, kT16);

  // flash attention (4 query tiles per block)
  flash_attn<<<dim3(T_SEQ / 64, NHEAD), 128, 0, stream>>>(q16, kT16, v16b, y16);

  // output projection
  gemm_f16_wmma<<<gD, 256, 0, stream>>>(y16, wo16, DMODEL, DMODEL, nullptr, attnF, nullptr, 0);

  // MLP
  dim3 gF(DFF / 128, T_SEQ / 128);      // (128, 16)
  gemm_f16_wmma<<<gF, 256, 0, stream>>>(h16, win16, DFF, DMODEL, b_in, nullptr, mh16, 1);
  gemm_f16_wmma<<<gD, 256, 0, stream>>>(mh16, wout16, DMODEL, DFF, b_out, mlpF, nullptr, 2);

  // residual
  final_add<<<4096, 256, 0, stream>>>(x, attnF, mlpF, out, TD);
}
